// GraphAttention_16243566314058
// MI455X (gfx1250) — matrix operand order repaired and checked by reading, not compiled, not run on hardware
//
#include <hip/hip_runtime.h>

typedef __bf16 bf16_t;
typedef __attribute__((ext_vector_type(16))) __bf16 v16bf;
typedef __attribute__((ext_vector_type(8)))  float  v8f;

#define GV   10000          // vocab
#define GVP  10016          // vocab padded to 32 (stage-2 B loads)
#define GE   128            // embed dim
#define GR   2048           // B*C rows
#define NREL 8
#define ROWS 16             // rows per workgroup
#define NBLK (GR / ROWS)    // 128 workgroups
#define NTHR 256            // 8 waves
#define NT16 ((GV + 15) / 16)   // 625 stage-1 v-tiles
#define NCH  ((GV + 31) / 32)   // 313 stage-2 v-chunks

__device__ __forceinline__ v8f wmma_bf16(v16bf a, v16bf b, v8f c) {
  return __builtin_amdgcn_wmma_f32_16x16x32_bf16(false, a, false, b, (short)0, c,
                                                 false, false);
}

// base tile: D[16 rows][16 v] = S_rows (16x128) x W^T (128x16), K in 4 chunks of 32.
__device__ __forceinline__ v8f base_tile(const v16bf a[4],
                                         const bf16_t* __restrict__ Wbf,
                                         int vvc, int lane) {
  v8f c = (v8f){0.f, 0.f, 0.f, 0.f, 0.f, 0.f, 0.f, 0.f};
  const bf16_t* row = Wbf + (size_t)vvc * GE + ((lane < 16) ? 0 : 16);
#pragma unroll
  for (int kc = 0; kc < 4; ++kc) {
    v16bf b = *(const v16bf*)(row + kc * 32);      // 32B contiguous, 32B aligned
    c = wmma_bf16(a[kc], b, c);
  }
  return c;
}

__global__ void prep_w(const float* __restrict__ W, bf16_t* __restrict__ Wbf,
                       bf16_t* __restrict__ Wtbf) {
  int idx = blockIdx.x * 256 + threadIdx.x;
  if (idx < GV * GE) {
    int v = idx / GE, e = idx % GE;
    float w = W[idx];
    Wbf[idx] = (bf16_t)w;
    Wtbf[(size_t)e * GVP + v] = (bf16_t)w;
  }
  if (idx < GE * (GVP - GV)) {                     // zero the transpose padding
    int e = idx / (GVP - GV), v = GV + idx % (GVP - GV);
    Wtbf[(size_t)e * GVP + v] = (bf16_t)0.f;
  }
}

__global__ void prep_colsum(const float* __restrict__ W, float* __restrict__ colsum) {
  int e = threadIdx.x;
  float s = 0.f;
  for (int v = 0; v < GV; ++v) s += W[(size_t)v * GE + e];
  colsum[e] = s;
}

__global__ __launch_bounds__(NTHR, 1)
void ga_fused(const int* __restrict__ src, const float* __restrict__ src_embed,
              const float* __restrict__ edge_mat, const int* __restrict__ rel_mat,
              const bf16_t* __restrict__ Wbf, const bf16_t* __restrict__ Wtbf,
              const float* __restrict__ colsum, float* __restrict__ num,
              unsigned char* __restrict__ cls_ws, float* __restrict__ out) {
  __shared__ float  Sf[ROWS * GE];
  __shared__ bf16_t Sb[ROWS * GE];
  __shared__ int    srcIdx[ROWS];
  __shared__ int    mBits[ROWS * NREL];            // per-(row,rel) max as float bits
  __shared__ float  mF[ROWS * NREL];
  __shared__ float  emF[ROWS * NREL];              // exp(-m) table
  __shared__ float  csL[ROWS * NREL];              // per-(row,rel) sum of c_v
  __shared__ float  cT[2][ROWS * 32];              // double-buffered coeff tile
  __shared__ unsigned char clsT[2][ROWS * 32];
  __shared__ float  scoreL[ROWS * NREL];
  __shared__ float  coefL[ROWS * NREL];
  __shared__ float  emL[ROWS * NREL];

  const int tid  = threadIdx.x;
  const int wave = tid >> 5;
  const int lane = tid & 31;
  const int nlo  = lane & 15;
  const int hi8  = (lane < 16) ? 0 : 8;
  const int r0   = blockIdx.x * ROWS;

  for (int i = tid; i < ROWS * GE; i += NTHR) {
    float s = src_embed[(size_t)r0 * GE + i];
    Sf[i] = s;
    Sb[i] = (bf16_t)s;
  }
  if (tid < ROWS) srcIdx[tid] = src[r0 + tid];
  if (tid < ROWS * NREL) { mBits[tid] = 0; csL[tid] = 0.f; }
  __syncthreads();

  // Per-lane row bases in registers (LDS reads once, not per tile).
  size_t gbase[8], cbase[8];
#pragma unroll
  for (int j = 0; j < 8; ++j) {
    int M = j + hi8;
    gbase[j] = (size_t)srcIdx[M] * GV;             // gather rows of rel/edge
    cbase[j] = (size_t)(r0 + M) * GV;              // class-cache rows
  }

  // Resident stage-1 A fragments (16-bit A layout).
  v16bf a[4];
#pragma unroll
  for (int kc = 0; kc < 4; ++kc)
#pragma unroll
    for (int j = 0; j < 16; ++j) {
      int k = kc * 32 + j + ((lane < 16) ? 0 : 16);
      a[kc][j] = Sb[nlo * GE + k];
    }

  // ---------------- pass 1: masked max + class-byte cache ----------------
  for (int tile = wave; tile < NT16; tile += 8) {
    int v0 = tile * 16;
    int vv = v0 + nlo;
    bool vok = vv < GV;
    int vvc = vok ? vv : GV - 1;
    // batched branchless gathers (compiler can clause these, overlap with WMMA)
    int   rel8[8];
    float ed8[8];
#pragma unroll
    for (int j = 0; j < 8; ++j) rel8[j] = rel_mat[gbase[j] + vvc];
#pragma unroll
    for (int j = 0; j < 8; ++j) ed8[j] = edge_mat[gbase[j] + vvc];
    v8f base = base_tile(a, Wbf, vvc, lane);
#pragma unroll
    for (int j = 0; j < 8; ++j) {
      int cls = (rel8[j] >= 1 && rel8[j] <= NREL && ed8[j] > 0.f) ? rel8[j] : 0;
      if (vok) cls_ws[cbase[j] + vv] = (unsigned char)cls;
      float bval = base[j];
      if (vok && cls && bval > 0.f)                // only >0 can raise max above init 0
        atomicMax(&mBits[(j + hi8) * NREL + cls - 1], __float_as_int(bval));
    }
  }
  __threadfence();                                 // cls_ws visible to all waves
  __syncthreads();
  if (tid < ROWS * NREL) {
    float m = __int_as_float(mBits[tid]);
    mF[tid] = m;
    emF[tid] = __expf(-m);
  }
  __syncthreads();

  // ---------------- pass 2: partitioned exp-weighted GEMM ----------------
  v8f acc[NREL];                                   // this wave owns E-tile [wave*16,+16)
#pragma unroll
  for (int i = 0; i < NREL; ++i)
    acc[i] = (v8f){0.f, 0.f, 0.f, 0.f, 0.f, 0.f, 0.f, 0.f};

  auto produce = [&](int ch) {                     // rotating producer pair
    if ((wave >> 1) == (ch & 3)) {
      int sub = wave & 1;
      int v0 = ch * 32 + sub * 16;
      int vv = v0 + nlo;
      bool vok = vv < GV;
      int vvc = vok ? vv : GV - 1;
      int bsel = ch & 1;
      int kk = sub * 16 + nlo;
      int cls8[8];
#pragma unroll
      for (int j = 0; j < 8; ++j)
        cls8[j] = vok ? (int)cls_ws[cbase[j] + vv] : 0;
      v8f base = base_tile(a, Wbf, vvc, lane);
#pragma unroll
      for (int j = 0; j < 8; ++j) {
        int M = j + hi8;
        float c = 0.f;
        int cls = cls8[j];
        if (cls) {
          int idx = M * NREL + cls - 1;
          c = __expf(base[j] - mF[idx]) - emF[idx];
          atomicAdd(&csL[idx], c);
        }
        cT[bsel][M * 32 + kk]   = c;
        clsT[bsel][M * 32 + kk] = (unsigned char)cls;
      }
    }
  };

  produce(0);
  __syncthreads();
  for (int ch = 0; ch < NCH; ++ch) {
    if (ch + 1 < NCH) produce(ch + 1);             // overlaps consume(ch)
    int bsel = ch & 1;
    // gather coeff row once (A layout), reuse across 8 relations
    bf16_t ckb[16];
    int clsk[16];
#pragma unroll
    for (int j = 0; j < 16; ++j) {
      int k = j + ((lane < 16) ? 0 : 16);
      ckb[j]  = (bf16_t)cT[bsel][nlo * 32 + k];
      clsk[j] = clsT[bsel][nlo * 32 + k];
    }
    const bf16_t* wt = Wtbf + (size_t)(wave * 16 + nlo) * GVP +
                       ch * 32 + ((lane < 16) ? 0 : 16);
    v16bf b0 = *(const v16bf*)wt;
#pragma unroll
    for (int i = 0; i < NREL; ++i) {
      v16bf af;
#pragma unroll
      for (int j = 0; j < 16; ++j)
        af[j] = (clsk[j] == i + 1) ? ckb[j] : (bf16_t)0.f;
      acc[i] = wmma_bf16(af, b0, acc[i]);
    }
    __syncthreads();
  }

  // spill per-relation numerators (wave owns its E-tile completely)
#pragma unroll
  for (int i = 0; i < NREL; ++i)
#pragma unroll
    for (int j = 0; j < 8; ++j) {
      int M = j + hi8;
      int e = wave * 16 + nlo;
      num[((size_t)(r0 + M) * NREL + i) * GE + e] = acc[i][j];
    }
  __threadfence();
  __syncthreads();

  // ---------------- finalize: Z, scores, 8-way softmax, mixture ----------------
  float Zloc = 1.f, scLoc = 0.f;
  if (tid < ROWS * NREL) {
    int M = tid >> 3, i = tid & 7;
    float em = emF[M * NREL + i];
    Zloc = (float)GV * em + csL[M * NREL + i];
    const float* nrow = num + ((size_t)(r0 + M) * NREL + i) * GE;
    float sc = 0.f;
    for (int e = 0; e < GE; ++e)
      sc += Sf[M * GE + e] * ((em * colsum[e] + nrow[e]) / Zloc);
    scLoc = sc;
    scoreL[tid] = sc;
    emL[tid] = em;
  }
  __syncthreads();
  if (tid < ROWS * NREL) {
    int M = tid >> 3;
    float mx = scoreL[M * NREL];
    for (int k = 1; k < NREL; ++k) mx = fmaxf(mx, scoreL[M * NREL + k]);
    float den = 0.f;
    for (int k = 0; k < NREL; ++k) den += __expf(scoreL[M * NREL + k] - mx);
    coefL[tid] = (__expf(scLoc - mx) / den) / Zloc;  // softmax weight / Z_i
  }
  __syncthreads();
  {
    int M = tid >> 4, col = tid & 15;
#pragma unroll
    for (int u = 0; u < 8; ++u) {
      int e = col + 16 * u;
      float o = 0.f;
#pragma unroll
      for (int k = 0; k < NREL; ++k)
        o += coefL[M * NREL + k] *
             (emL[M * NREL + k] * colsum[e] +
              num[((size_t)(r0 + M) * NREL + k) * GE + e]);
      out[(size_t)(r0 + M) * GE + e] = o;
    }
  }
}

extern "C" void kernel_launch(void* const* d_in, const int* in_sizes, int n_in,
                              void* d_out, int out_size, void* d_ws, size_t ws_size,
                              hipStream_t stream) {
  (void)in_sizes; (void)n_in; (void)out_size; (void)ws_size;
  const int*   src       = (const int*)  d_in[0];
  const float* src_embed = (const float*)d_in[1];
  const float* W         = (const float*)d_in[2];
  const float* edge      = (const float*)d_in[3];
  const int*   rel       = (const int*)  d_in[4];

  char* ws = (char*)d_ws;
  size_t off = 0;
  auto take = [&](size_t bytes) {
    void* p = ws + off;
    off = (off + bytes + 255) & ~(size_t)255;
    return p;
  };
  bf16_t* Wbf    = (bf16_t*)take((size_t)GV * GE * sizeof(bf16_t));       // 2.56 MB
  bf16_t* Wtbf   = (bf16_t*)take((size_t)GE * GVP * sizeof(bf16_t));      // 2.56 MB
  float*  colsum = (float*) take((size_t)GE * sizeof(float));
  float*  num    = (float*) take((size_t)GR * NREL * GE * sizeof(float)); // 8.4 MB
  unsigned char* clsw = (unsigned char*)take((size_t)GR * GV);            // 20.5 MB

  prep_w<<<(GV * GE + 255) / 256, 256, 0, stream>>>(W, Wbf, Wtbf);
  prep_colsum<<<1, GE, 0, stream>>>(W, colsum);
  ga_fused<<<NBLK, NTHR, 0, stream>>>(src, src_embed, edge, rel, Wbf, Wtbf, colsum,
                                      num, clsw, (float*)d_out);
}
